// GPTLanguageModel_16965120819564
// MI455X (gfx1250) — matrix operand order repaired and checked by reading, compiled, not run on hardware
//
#include <hip/hip_runtime.h>
#include <cstdint>
#include <cstddef>

// ---------------------------------------------------------------------------
// GPT forward (B=32, T=256, D=384, H=6, HD=64, L=6, V=32000) for gfx1250.
// GEMMs: v_wmma_f32_16x16x32_bf16, double-buffered LDS fed by
// global_load_async_to_lds_b128 (ASYNCcnt), weights pre-transposed to [N,K].
// ---------------------------------------------------------------------------

typedef __bf16 bf16;
typedef __attribute__((ext_vector_type(16))) __bf16 v16bf;
typedef __attribute__((ext_vector_type(8)))  float  v8f;

#define V_SZ 32000
#define D_SZ 384
#define T_SZ 256
#define H_SZ 6
#define HD_SZ 64
#define L_SZ 6
#define B_SZ 32
#define MROWS (B_SZ * T_SZ)   // 8192
#define FF_SZ (4 * D_SZ)      // 1536

// ------------------------------ GEMM -----------------------------------
// C[M,N] = A[M,K](bf16 row-major) * Bt[N,K](bf16, pre-transposed)
// [+bias] [+resid] [relu]; out fp32 (Cf) or bf16 (Cb).
// Block 128x128, K-step 32, 8 waves; wave tile 64x32 = 4x2 WMMA tiles.
#define BM 128
#define BN 128
#define BK 32
#define BKP 40     // padded LDS pitch (elements): 80B rows -> 16B aligned segs

__device__ __forceinline__ v16bf frag_a(const bf16* row, int half) {
    // A 16x32 layout: pairs j=0..3 -> K = half*8 + 2j ; j=4..7 -> K = 16+half*8+2(j-4)
    union { uint4 u[2]; v16bf v; } t;
    t.u[0] = *(const uint4*)(row + half * 8);
    t.u[1] = *(const uint4*)(row + 16 + half * 8);
    return t.v;
}
__device__ __forceinline__ v16bf frag_b(const bf16* col, int half) {
    // B 32x16 layout: lane = N col, K = half*8 + 0..7, then 16 + half*8 + 0..7 (matches A)
    union { uint4 u[2]; v16bf v; } t;
    t.u[0] = *(const uint4*)(col + half * 8);
    t.u[1] = *(const uint4*)(col + 16 + half * 8);
    return t.v;
}

__global__ __launch_bounds__(256) void gemm_bf16_wmma(
    const bf16* __restrict__ A, const bf16* __restrict__ Bt,
    const float* __restrict__ bias, const float* __restrict__ resid,
    float* __restrict__ Cf, bf16* __restrict__ Cb,
    int M, int N, int K, int relu)
{
    __shared__ __align__(16) bf16 As[2][BM][BKP];   // 2 x 10 KB
    __shared__ __align__(16) bf16 Bs[2][BN][BKP];   // 2 x 10 KB

    const int tid  = threadIdx.x;
    const int wave = tid >> 5;
    const int lane = tid & 31;
    const int wm2  = wave >> 2;         // 0..1 : 64-row slice
    const int wn2  = wave & 3;          // 0..3 : 32-col slice
    const int half = lane >> 4;
    const int lh   = lane & 15;
    const int mBase = blockIdx.y * BM;
    const int nBase = blockIdx.x * BN;

    // issue 4 async b128 loads per thread: 128 rows x 32 K for A and Bt tiles
    auto stage = [&](int kb, int buf) {
        #pragma unroll
        for (int i = 0; i < 2; ++i) {
            int cc  = tid + 256 * i;        // 0..511
            int row = cc >> 2;              // 0..127
            int cs  = (cc & 3) << 3;        // 0,8,16,24 (elements)
            uint32_t la = (uint32_t)(uintptr_t)&As[buf][row][cs];
            uint64_t ga = (uint64_t)(uintptr_t)(A + (size_t)(mBase + row) * K + kb + cs);
            asm volatile("global_load_async_to_lds_b128 %0, %1, off"
                         :: "v"(la), "v"(ga) : "memory");
            uint32_t lb = (uint32_t)(uintptr_t)&Bs[buf][row][cs];
            uint64_t gb = (uint64_t)(uintptr_t)(Bt + (size_t)(nBase + row) * K + kb + cs);
            asm volatile("global_load_async_to_lds_b128 %0, %1, off"
                         :: "v"(lb), "v"(gb) : "memory");
        }
    };

    v8f acc[4][2] = {};
    const int nsteps = K / BK;

    stage(0, 0);
    asm volatile("s_wait_asynccnt 0x0" ::: "memory");
    __syncthreads();

    for (int s = 0; s < nsteps; ++s) {
        const int buf = s & 1;
        if (s + 1 < nsteps) stage((s + 1) * BK, buf ^ 1);   // prefetch next tile

        v16bf bq0 = frag_b(&Bs[buf][wn2 * 32 + lh][0],      half);
        v16bf bq1 = frag_b(&Bs[buf][wn2 * 32 + 16 + lh][0], half);
        #pragma unroll
        for (int i = 0; i < 4; ++i) {
            v16bf af = frag_a(&As[buf][wm2 * 64 + i * 16 + lh][0], half);
            acc[i][0] = __builtin_amdgcn_wmma_f32_16x16x32_bf16(
                false, af, false, bq0, (short)0, acc[i][0], false, false);
            acc[i][1] = __builtin_amdgcn_wmma_f32_16x16x32_bf16(
                false, af, false, bq1, (short)0, acc[i][1], false, false);
        }
        asm volatile("s_wait_asynccnt 0x0" ::: "memory");
        __syncthreads();
    }

    // ---- epilogue: bias / residual / relu, fp32 or bf16 store ----
    #pragma unroll
    for (int i = 0; i < 4; ++i) {
        #pragma unroll
        for (int j = 0; j < 2; ++j) {
            int col = nBase + wn2 * 32 + j * 16 + lh;
            float bv = bias ? bias[col] : 0.f;
            #pragma unroll
            for (int r = 0; r < 8; ++r) {
                int row = mBase + wm2 * 64 + i * 16 + r + half * 8;
                float cv = acc[i][j][r] + bv;
                if (resid) cv += resid[(size_t)row * N + col];
                if (relu)  cv = fmaxf(cv, 0.f);
                if (Cf) Cf[(size_t)row * N + col] = cv;
                else    Cb[(size_t)row * N + col] = (bf16)cv;
            }
        }
    }
}

// --------------------------- Embedding ---------------------------------
__global__ void embed_kernel(const int* __restrict__ idx,
                             const float* __restrict__ tok,
                             const float* __restrict__ pos,
                             float* __restrict__ x)
{
    int gid = blockIdx.x * blockDim.x + threadIdx.x;
    if (gid >= MROWS * D_SZ) return;
    int d  = gid % D_SZ;
    int bt = gid / D_SZ;
    int t  = bt % T_SZ;
    x[gid] = tok[(size_t)idx[bt] * D_SZ + d] + pos[t * D_SZ + d];
}

// --------------------------- LayerNorm ---------------------------------
__global__ __launch_bounds__(128) void ln_kernel(
    const float* __restrict__ x, const float* __restrict__ g,
    const float* __restrict__ b, bf16* __restrict__ out)
{
    __shared__ float red[128];
    const int row = blockIdx.x;
    const int tid = threadIdx.x;
    const float* xr = x + (size_t)row * D_SZ;
    float v0 = xr[tid], v1 = xr[tid + 128], v2 = xr[tid + 256];
    red[tid] = v0 + v1 + v2;
    __syncthreads();
    for (int off = 64; off > 0; off >>= 1) {
        if (tid < off) red[tid] += red[tid + off];
        __syncthreads();
    }
    float mu = red[0] * (1.f / D_SZ);
    __syncthreads();
    float d0 = v0 - mu, d1 = v1 - mu, d2 = v2 - mu;
    red[tid] = d0 * d0 + d1 * d1 + d2 * d2;
    __syncthreads();
    for (int off = 64; off > 0; off >>= 1) {
        if (tid < off) red[tid] += red[tid + off];
        __syncthreads();
    }
    float rstd = rsqrtf(red[0] * (1.f / D_SZ) + 1e-5f);
    bf16* orow = out + (size_t)row * D_SZ;
    orow[tid]       = (bf16)(d0 * rstd * g[tid]       + b[tid]);
    orow[tid + 128] = (bf16)(d1 * rstd * g[tid + 128] + b[tid + 128]);
    orow[tid + 256] = (bf16)(d2 * rstd * g[tid + 256] + b[tid + 256]);
}

// ----------------------- Weight conversions ----------------------------
// Tiled transpose: in [l][K][N] fp32 -> out [l][N][K] bf16. Block (32,8).
__global__ __launch_bounds__(256) void transpose_f2b(
    const float* __restrict__ in, bf16* __restrict__ out, int K, int N)
{
    __shared__ float tile[32][33];
    const int l  = blockIdx.z;
    const int n0 = blockIdx.x * 32, k0 = blockIdx.y * 32;
    const int tx = threadIdx.x, ty = threadIdx.y;
    in  += (size_t)l * K * N;
    out += (size_t)l * N * K;
    #pragma unroll
    for (int r = ty; r < 32; r += 8)
        tile[r][tx] = in[(size_t)(k0 + r) * N + n0 + tx];
    __syncthreads();
    #pragma unroll
    for (int r = ty; r < 32; r += 8)
        out[(size_t)(n0 + r) * K + k0 + tx] = (bf16)tile[tx][r];
}

// [L,H,D,HD] fp32 -> [L, N=H*HD, K=D] bf16 (head-merged + transposed)
__global__ void trans_qkv_kernel(const float* __restrict__ in,
                                 bf16* __restrict__ out, int n)
{
    int gid = blockIdx.x * blockDim.x + threadIdx.x;
    if (gid >= n) return;
    int k  = gid % D_SZ;
    int nl = gid / D_SZ;
    int nn = nl % D_SZ;            // h*64 + hd
    int l  = nl / D_SZ;
    int h  = nn >> 6, hd = nn & 63;
    out[gid] = (bf16)in[(((size_t)l * H_SZ + h) * D_SZ + k) * HD_SZ + hd];
}

// --------------------------- Attention ---------------------------------
// q,k,v bf16 in [B,T,H,HD]; one block per (b,h); thread = query row t.
__global__ __launch_bounds__(256) void attn_kernel(
    const bf16* __restrict__ q, const bf16* __restrict__ k,
    const bf16* __restrict__ v, bf16* __restrict__ o)
{
    __shared__ bf16 ks[T_SZ][HD_SZ];
    __shared__ bf16 vs[T_SZ][HD_SZ];
    const int bh = blockIdx.x;
    const int b  = bh / H_SZ;
    const int h  = bh % H_SZ;
    const int t  = threadIdx.x;

    const bf16* kbase = k + (size_t)b * T_SZ * D_SZ + h * HD_SZ;
    const bf16* vbase = v + (size_t)b * T_SZ * D_SZ + h * HD_SZ;
    #pragma unroll
    for (int d = 0; d < HD_SZ; d += 8) {
        *(uint4*)&ks[t][d] = *(const uint4*)(kbase + (size_t)t * D_SZ + d);
        *(uint4*)&vs[t][d] = *(const uint4*)(vbase + (size_t)t * D_SZ + d);
    }
    __syncthreads();

    const float scale = 0.125f;     // 1/sqrt(64)
    float qr[HD_SZ];
    const bf16* qrow = q + ((size_t)(b * T_SZ + t)) * D_SZ + h * HD_SZ;
    #pragma unroll
    for (int d = 0; d < HD_SZ; ++d) qr[d] = (float)qrow[d] * scale;

    float m = -1e30f, lsum = 0.f;
    float oacc[HD_SZ];
    #pragma unroll
    for (int d = 0; d < HD_SZ; ++d) oacc[d] = 0.f;

    for (int s = 0; s <= t; ++s) {
        float sc = 0.f;
        #pragma unroll
        for (int d = 0; d < HD_SZ; ++d) sc += qr[d] * (float)ks[s][d];
        float mn   = fmaxf(m, sc);
        float corr = __expf(m - mn);
        float p    = __expf(sc - mn);
        lsum = lsum * corr + p;
        #pragma unroll
        for (int d = 0; d < HD_SZ; ++d)
            oacc[d] = oacc[d] * corr + p * (float)vs[s][d];
        m = mn;
    }
    float inv = 1.f / lsum;
    bf16* orow = o + ((size_t)(b * T_SZ + t)) * D_SZ + h * HD_SZ;
    #pragma unroll
    for (int d = 0; d < HD_SZ; ++d) orow[d] = (bf16)(oacc[d] * inv);
}

// --------------------------- Host side ---------------------------------
static inline void run_gemm(const bf16* A, const bf16* Bt, const float* bias,
                            const float* resid, float* Cf, bf16* Cb,
                            int M, int N, int K, int relu, hipStream_t s)
{
    dim3 grid(N / BN, M / BM);
    gemm_bf16_wmma<<<grid, 256, 0, s>>>(A, Bt, bias, resid, Cf, Cb, M, N, K, relu);
}

extern "C" void kernel_launch(void* const* d_in, const int* in_sizes, int n_in,
                              void* d_out, int out_size, void* d_ws, size_t ws_size,
                              hipStream_t stream)
{
    (void)in_sizes; (void)n_in; (void)out_size;

    const int*   idx   = (const int*)  d_in[0];
    const float* tok   = (const float*)d_in[1];
    const float* pos   = (const float*)d_in[2];
    const float* wq    = (const float*)d_in[3];
    const float* wk    = (const float*)d_in[4];
    const float* wv    = (const float*)d_in[5];
    const float* wo    = (const float*)d_in[6];
    const float* bo    = (const float*)d_in[7];
    const float* ln1g  = (const float*)d_in[8];
    const float* ln1b  = (const float*)d_in[9];
    const float* ln2g  = (const float*)d_in[10];
    const float* ln2b  = (const float*)d_in[11];
    const float* w1    = (const float*)d_in[12];
    const float* b1    = (const float*)d_in[13];
    const float* w2    = (const float*)d_in[14];
    const float* b2    = (const float*)d_in[15];
    const float* lnfg  = (const float*)d_in[16];
    const float* lnfb  = (const float*)d_in[17];
    const float* wf    = (const float*)d_in[18];
    const float* bfv   = (const float*)d_in[19];
    float* logits      = (float*)d_out;

    // ---- workspace layout (all offsets 256B aligned) ----
    char* ws = (char*)d_ws;
    size_t off = 0;
    auto take = [&](size_t bytes) { char* p = ws + off; off += bytes; return p; };
    float* x      = (float*)take((size_t)MROWS * D_SZ * 4);
    bf16*  h_bf   = (bf16*) take((size_t)MROWS * D_SZ * 2);
    bf16*  q_bf   = (bf16*) take((size_t)MROWS * D_SZ * 2);
    bf16*  k_bf   = (bf16*) take((size_t)MROWS * D_SZ * 2);
    bf16*  v_bf   = (bf16*) take((size_t)MROWS * D_SZ * 2);
    bf16*  o_bf   = (bf16*) take((size_t)MROWS * D_SZ * 2);
    bf16*  mid_bf = (bf16*) take((size_t)MROWS * FF_SZ * 2);
    bf16*  wq_bf  = (bf16*) take((size_t)L_SZ * D_SZ * D_SZ * 2);   // [N,K]
    bf16*  wk_bf  = (bf16*) take((size_t)L_SZ * D_SZ * D_SZ * 2);
    bf16*  wv_bf  = (bf16*) take((size_t)L_SZ * D_SZ * D_SZ * 2);
    bf16*  wo_bf  = (bf16*) take((size_t)L_SZ * D_SZ * D_SZ * 2);
    bf16*  w1_bf  = (bf16*) take((size_t)L_SZ * FF_SZ * D_SZ * 2);  // [N=1536,K=384]
    bf16*  w2_bf  = (bf16*) take((size_t)L_SZ * D_SZ * FF_SZ * 2);  // [N=384,K=1536]
    bf16*  wf_bf  = (bf16*) take((size_t)V_SZ * D_SZ * 2);          // [N=32000,K=384]
    if (off > ws_size) return;   // deterministic no-op on undersized ws

    // ---- weight conversion: fp32 -> bf16, transposed to [N,K] ----
    {
        int nq = L_SZ * D_SZ * D_SZ;
        int blocks = (nq + 255) / 256;
        trans_qkv_kernel<<<blocks, 256, 0, stream>>>(wq, wq_bf, nq);
        trans_qkv_kernel<<<blocks, 256, 0, stream>>>(wk, wk_bf, nq);
        trans_qkv_kernel<<<blocks, 256, 0, stream>>>(wv, wv_bf, nq);
        dim3 tb(32, 8);
        transpose_f2b<<<dim3(D_SZ / 32, D_SZ / 32, L_SZ), tb, 0, stream>>>(
            wo, wo_bf, D_SZ, D_SZ);
        transpose_f2b<<<dim3(FF_SZ / 32, D_SZ / 32, L_SZ), tb, 0, stream>>>(
            w1, w1_bf, D_SZ, FF_SZ);
        transpose_f2b<<<dim3(D_SZ / 32, FF_SZ / 32, L_SZ), tb, 0, stream>>>(
            w2, w2_bf, FF_SZ, D_SZ);
        transpose_f2b<<<dim3(V_SZ / 32, D_SZ / 32, 1), tb, 0, stream>>>(
            wf, wf_bf, D_SZ, V_SZ);
    }

    // ---- embedding ----
    embed_kernel<<<(MROWS * D_SZ + 255) / 256, 256, 0, stream>>>(idx, tok, pos, x);

    // ---- transformer layers ----
    for (int l = 0; l < L_SZ; ++l) {
        const bf16* wql = wq_bf + (size_t)l * D_SZ * D_SZ;
        const bf16* wkl = wk_bf + (size_t)l * D_SZ * D_SZ;
        const bf16* wvl = wv_bf + (size_t)l * D_SZ * D_SZ;
        const bf16* wol = wo_bf + (size_t)l * D_SZ * D_SZ;
        const bf16* w1l = w1_bf + (size_t)l * D_SZ * FF_SZ;
        const bf16* w2l = w2_bf + (size_t)l * FF_SZ * D_SZ;

        ln_kernel<<<MROWS, 128, 0, stream>>>(x, ln1g + l * D_SZ, ln1b + l * D_SZ, h_bf);

        run_gemm(h_bf, wql, nullptr, nullptr, nullptr, q_bf,
                 MROWS, D_SZ, D_SZ, 0, stream);
        run_gemm(h_bf, wkl, nullptr, nullptr, nullptr, k_bf,
                 MROWS, D_SZ, D_SZ, 0, stream);
        run_gemm(h_bf, wvl, nullptr, nullptr, nullptr, v_bf,
                 MROWS, D_SZ, D_SZ, 0, stream);

        attn_kernel<<<B_SZ * H_SZ, T_SZ, 0, stream>>>(q_bf, k_bf, v_bf, o_bf);

        // x = x + o @ wo + bo
        run_gemm(o_bf, wol, bo + l * D_SZ, x, x, nullptr,
                 MROWS, D_SZ, D_SZ, 0, stream);

        ln_kernel<<<MROWS, 128, 0, stream>>>(x, ln2g + l * D_SZ, ln2b + l * D_SZ, h_bf);

        // mid = relu(h @ w1 + b1)  (bf16 out)
        run_gemm(h_bf, w1l, b1 + l * FF_SZ, nullptr, nullptr, mid_bf,
                 MROWS, FF_SZ, D_SZ, 1, stream);
        // x = x + mid @ w2 + b2
        run_gemm(mid_bf, w2l, b2 + l * D_SZ, x, x, nullptr,
                 MROWS, D_SZ, FF_SZ, 0, stream);
    }

    // ---- final LN + logits ----
    ln_kernel<<<MROWS, 128, 0, stream>>>(x, lnfg, lnfb, h_bf);
    run_gemm(h_bf, wf_bf, bfv, nullptr, logits, nullptr,
             MROWS, V_SZ, D_SZ, 0, stream);
}
